// AudioCodebook_17927193493856
// MI455X (gfx1250) — matrix operand order repaired and checked by reading, not compiled, not run on hardware
//
#include <hip/hip_runtime.h>
#include <hip/hip_bf16.h>

// ---------------- problem constants ----------------
#define D       1024
#define K       8192
#define NTOK    32768          // B*S = 16*2048
#define DECAY   0.99f
#define ONE_M_D 0.01f          // 1 - DECAY
#define EPSV    1e-05f
#define COMMIT  0.25f

// ---------------- phase-1 tiling ----------------
#define MT      64             // tokens per workgroup
#define CT      128            // codes per tile
#define DT      32             // feature depth per WMMA step
#define ZSTRIDE 1032           // padded LDS row stride (bf16 elems) for z
#define CSTRIDE 56             // padded LDS row stride (bf16 elems) for code tile
#define TILEHW  (CT * CSTRIDE) // one (hi or lo) plane in LDS, bf16 elems
#define TILEW   (2 * TILEHW)   // hi+lo planes per buffer

typedef __attribute__((ext_vector_type(16))) __bf16 v16bf;
typedef __attribute__((ext_vector_type(8)))  float  v8f;
typedef __attribute__((ext_vector_type(4)))  unsigned int v4u;
typedef __attribute__((ext_vector_type(8)))  int v8i;
typedef __attribute__((ext_vector_type(4)))  int v4i;

struct __align__(8) U16x4 { unsigned short x, y, z, w; };

union AF { uint4 q[2]; v16bf v; };

static __device__ __forceinline__ v8f wmma_bf16(v16bf a, v16bf b, v8f c) {
    return __builtin_amdgcn_wmma_f32_16x16x32_bf16(
        /*neg_a*/false, a, /*neg_b*/false, b,
        /*c_mod*/(short)0, c, /*reuse_a*/false, /*reuse_b*/false);
}

static __device__ __forceinline__ void split_bf16(float f, unsigned short& h, unsigned short& l) {
    unsigned u  = __float_as_uint(f);
    unsigned hb = (u + 0x7FFFu + ((u >> 16) & 1u)) >> 16;     // RNE to bf16
    float hf    = __uint_as_float(hb << 16);
    float lo    = f - hf;
    unsigned ul = __float_as_uint(lo);
    unsigned lb = (ul + 0x7FFFu + ((ul >> 16) & 1u)) >> 16;
    h = (unsigned short)hb;
    l = (unsigned short)lb;
}

static __device__ __forceinline__ unsigned long long umin64(unsigned long long a,
                                                            unsigned long long b) {
    return a < b ? a : b;
}

// monotonic key: order of key equals order of float
static __device__ __forceinline__ unsigned fkey(float f) {
    unsigned u = __float_as_uint(f);
    return (u & 0x80000000u) ? ~u : (u | 0x80000000u);
}

// =====================================================================
// TDM: one 3-D descriptor moves a (hi+lo) code tile, 2 x 128 x 32 bf16,
// from the [2][K][D] split-codebook into LDS, injecting 48B of padding
// after every 64B row so LDS rows land at the conflict-free 112B stride.
//   x = DT features (contiguous), y = CT codes (stride D),
//   z = 2 planes (stride K*D).
// =====================================================================
static __device__ __forceinline__ void tdm_load_tile(const unsigned short* gbase,
                                                     unsigned lds_off,
                                                     int ct0, int d0) {
    unsigned long long ga = (unsigned long long)gbase
                          + 2ull * ((unsigned long long)ct0 * D + (unsigned)d0);
    v4u g0 = { 1u,                                   // count=1, load descriptor
               lds_off,                              // LDS destination (bytes)
               (unsigned)(ga & 0xFFFFFFFFull),
               (unsigned)((ga >> 32) & 0x1FFFFFFull) | (2u << 30) };  // type=2
    v8i g1 = { (int)((1u << 16)      // data_size = 2 bytes
                   | (1u << 20)      // pad_enable
                   | (3u << 22)      // pad_interval: every 16 DWORDs (64B row)
                   | (11u << 25)),   // pad_amount: 12 DWORDs (48B)
               (int)(1024u << 16),                  // tensor_dim0 = D
               (int)(8192u << 16),                  // tensor_dim1 = K
               (int)(32u << 16),                    // tile_dim0 = DT
               (int)(128u | (2u << 16)),            // tile_dim1 = CT, tile_dim2 = 2
               1024,                                // tensor_dim0_stride = D
               0,                                   // stride0_hi | stride1_lo
               0x80 };                              // tensor_dim1_stride = K*D (>>16)
    v4i g2 = { 2, 0, 0, 0 };                        // tensor_dim2 = 2
    v4i g3 = { 0, 0, 0, 0 };
#if __clang_major__ >= 23
    v8i g4 = { 0, 0, 0, 0, 0, 0, 0, 0 };
    __builtin_amdgcn_tensor_load_to_lds(g0, g1, g2, g3, g4, 0);
#else
    __builtin_amdgcn_tensor_load_to_lds(g0, g1, g2, g3, 0);
#endif
}

// =====================================================================
// K0a: codebook prep: fp32 -> bf16 hi/lo planes ([2][K][D]) + row ||c||^2
// grid: K blocks x 256 threads
// =====================================================================
__global__ __launch_bounds__(256) void k_prep_cb(const float* __restrict__ cb,
                                                 unsigned short* __restrict__ c2,
                                                 float* __restrict__ cnorm) {
    __shared__ float red[256];
    const int k = blockIdx.x;
    const int t = threadIdx.x;
    float4 c4 = ((const float4*)(cb + (size_t)k * D))[t];
    U16x4 h4, l4;
    split_bf16(c4.x, h4.x, l4.x);
    split_bf16(c4.y, h4.y, l4.y);
    split_bf16(c4.z, h4.z, l4.z);
    split_bf16(c4.w, h4.w, l4.w);
    *(U16x4*)(c2 + (size_t)k * D + t * 4)                 = h4;   // hi plane
    *(U16x4*)(c2 + (size_t)K * D + (size_t)k * D + t * 4) = l4;   // lo plane
    float s = c4.x * c4.x + c4.y * c4.y + c4.z * c4.z + c4.w * c4.w;
    red[t] = s;
    __syncthreads();
    for (int o = 128; o > 0; o >>= 1) {
        if (t < o) red[t] += red[t + o];
        __syncthreads();
    }
    if (t == 0) cnorm[k] = red[0];
}

// =====================================================================
// K0b: zero scratch
// =====================================================================
__global__ void k_zero(float* __restrict__ p, int n) {
    int i = blockIdx.x * 256 + threadIdx.x;
    if (i < n) p[i] = 0.0f;
}

// =====================================================================
// K1: fused distance-GEMM + argmin.   grid: NTOK/MT blocks x 256 threads
// bf16 hi/lo (3-term) WMMA emulation of fp32  z . C^T ; running argmin
// of  ||c||^2 - 2 z.c  per token.  Code tiles stream in via TDM with
// double-buffered LDS, synchronized by TENSORcnt + split barriers.
// =====================================================================
__global__ __launch_bounds__(256, 1) void k_argmin(const float* __restrict__ z,
                                                   const unsigned short* __restrict__ c2,
                                                   const float* __restrict__ cnorm,
                                                   int* __restrict__ ids,
                                                   float* __restrict__ out_ids) {
    __shared__ __align__(16) unsigned short s_zhi[MT * ZSTRIDE];
    __shared__ __align__(16) unsigned short s_zlo[MT * ZSTRIDE];
    __shared__ __align__(16) unsigned short s_tile[2 * TILEW];   // double buffer
    __shared__ unsigned long long s_minpk[MT];

    const int t    = threadIdx.x;
    const int tok0 = blockIdx.x * MT;
    const int wid  = __builtin_amdgcn_readfirstlane(t >> 5);     // scalar wave id

    // ---- wave 0 kicks off the first TDM tile before staging z ----
    if (wid == 0) {
        tdm_load_tile(c2, (unsigned)(unsigned long long)(size_t)&s_tile[0], 0, 0);
    }

    // ---- stage token block: fp32 -> bf16 hi/lo into LDS (padded rows) ----
    for (int i = t; i < MT * D / 4; i += 256) {
        int e = i * 4;
        int row = e >> 10;          // /D
        int col = e & (D - 1);
        float4 v = *(const float4*)(z + (size_t)(tok0 + row) * D + col);
        U16x4 h4, l4;
        split_bf16(v.x, h4.x, l4.x);
        split_bf16(v.y, h4.y, l4.y);
        split_bf16(v.z, h4.z, l4.z);
        split_bf16(v.w, h4.w, l4.w);
        *(U16x4*)(s_zhi + row * ZSTRIDE + col) = h4;
        *(U16x4*)(s_zlo + row * ZSTRIDE + col) = l4;
    }
    if (t < MT) s_minpk[t] = ~0ull;

    // ---- wave layout: 8 waves as 4(M) x 2(N); each wave: 16 rows x 64 codes ----
    const int lane = t & 31;
    const int w    = t >> 5;
    const int m0   = (w >> 1) * 16;
    const int n0   = (w & 1) * 64;
    const int lm   = lane & 15;
    const int lg   = lane >> 4;

    v8f acc[4];
#pragma unroll
    for (int st = 0; st < 4; ++st) acc[st] = (v8f){0, 0, 0, 0, 0, 0, 0, 0};

    const int NSTEPS = (K / CT) * (D / DT);   // 64 * 32 = 2048
    for (int s = 0; s < NSTEPS; ++s) {
        const int buf = s & 1;
        __syncthreads();                      // buffer buf^1 free for tile s+1
        if (wid == 0) {
            if (s + 1 < NSTEPS) {
                int ns  = s + 1;
                int nct = (ns >> 5) * CT;
                int nd0 = (ns & 31) * DT;
                unsigned loff = (unsigned)(unsigned long long)(size_t)
                                    &s_tile[(buf ^ 1) * TILEW];
                tdm_load_tile(c2, loff, nct, nd0);
                __builtin_amdgcn_s_wait_tensorcnt(1);   // tile s has landed
            } else {
                __builtin_amdgcn_s_wait_tensorcnt(0);
            }
        }
        __syncthreads();                      // tile s visible to all waves

        const unsigned short* sc_hi = s_tile + buf * TILEW;
        const unsigned short* sc_lo = sc_hi + TILEHW;

        // ---- A fragments (16x32 bf16, ISA lane layout) ----
        const int d0 = (s & 31) * DT;
        AF ah, al;
        const int ab = (m0 + lm) * ZSTRIDE + d0 + lg * 8;
        ah.q[0] = *(const uint4*)(s_zhi + ab);
        ah.q[1] = *(const uint4*)(s_zhi + ab + 16);
        al.q[0] = *(const uint4*)(s_zlo + ab);
        al.q[1] = *(const uint4*)(s_zlo + ab + 16);

#pragma unroll
        for (int st = 0; st < 4; ++st) {
            AF bh, bl;
            const int bb = (n0 + st * 16 + lm) * CSTRIDE + lg * 8;
            bh.q[0] = *(const uint4*)(sc_hi + bb);
            bh.q[1] = *(const uint4*)(sc_hi + bb + 16);
            bl.q[0] = *(const uint4*)(sc_lo + bb);
            bl.q[1] = *(const uint4*)(sc_lo + bb + 16);
            acc[st] = wmma_bf16(ah.v, bh.v, acc[st]);   // hi*hi
            acc[st] = wmma_bf16(ah.v, bl.v, acc[st]);   // hi*lo
            acc[st] = wmma_bf16(al.v, bh.v, acc[st]);   // lo*hi
        }

        // ---- end of a code tile: fold scores into running argmin ----
        if ((s & 31) == 31) {
            const int ct0 = (s >> 5) * CT;
            unsigned long long best[8];
#pragma unroll
            for (int r = 0; r < 8; ++r) best[r] = ~0ull;
#pragma unroll
            for (int st = 0; st < 4; ++st) {
                const int code = ct0 + n0 + st * 16 + lm;
                const float cn = cnorm[code];
#pragma unroll
                for (int r = 0; r < 8; ++r) {
                    float dist = cn - 2.0f * acc[st][r];
                    unsigned long long p =
                        ((unsigned long long)fkey(dist) << 32) | (unsigned)code;
                    best[r] = umin64(best[r], p);
                }
            }
#pragma unroll
            for (int r = 0; r < 8; ++r) {       // reduce across 16 lanes of half-wave
#pragma unroll
                for (int off = 8; off >= 1; off >>= 1) {
                    unsigned long long o = __shfl_xor(best[r], off, 16);
                    best[r] = umin64(best[r], o);
                }
            }
            if (lm == 0) {
#pragma unroll
                for (int r = 0; r < 8; ++r)
                    atomicMin(&s_minpk[m0 + lg * 8 + r], best[r]);
            }
#pragma unroll
            for (int st = 0; st < 4; ++st) acc[st] = (v8f){0, 0, 0, 0, 0, 0, 0, 0};
        }
    }

    __syncthreads();
    if (t < MT) {
        unsigned long long p = s_minpk[t];
        int id = (int)(unsigned)(p & 0xFFFFFFFFull);
        ids[tok0 + t]     = id;
        out_ids[tok0 + t] = (float)id;
    }
}

// =====================================================================
// K2: gather z_q, commitment-loss partial, cluster histogram, dw scatter
// grid: NTOK blocks x 256 threads (one token per block)
// =====================================================================
__global__ __launch_bounds__(256) void k_gather(const float* __restrict__ z,
                                                const float* __restrict__ cb,
                                                const int* __restrict__ ids,
                                                float* __restrict__ out_zq,
                                                float* __restrict__ dw,
                                                float* __restrict__ cs,
                                                float* __restrict__ loss_acc) {
    __shared__ float red[256];
    const int n = blockIdx.x;
    const int t = threadIdx.x;
    const int id = ids[n];
    float4 zv = ((const float4*)(z  + (size_t)n  * D))[t];
    float4 cv = ((const float4*)(cb + (size_t)id * D))[t];
    ((float4*)(out_zq + (size_t)n * D))[t] = cv;       // STE value == gathered code
    float dx = cv.x - zv.x, dy = cv.y - zv.y, dz = cv.z - zv.z, dww = cv.w - zv.w;
    red[t] = dx * dx + dy * dy + dz * dz + dww * dww;
    float* dp = dw + (size_t)id * D + t * 4;           // coalesced fp32 atomics
    atomicAdd(dp + 0, zv.x);
    atomicAdd(dp + 1, zv.y);
    atomicAdd(dp + 2, zv.z);
    atomicAdd(dp + 3, zv.w);
    __syncthreads();
    for (int o = 128; o > 0; o >>= 1) {
        if (t < o) red[t] += red[t + o];
        __syncthreads();
    }
    if (t == 0) {
        atomicAdd(loss_acc, red[0]);
        atomicAdd(cs + id, 1.0f);
    }
}

// =====================================================================
// K3: new_ema_cluster_size + global sum n.   grid: K/256 blocks
// =====================================================================
__global__ __launch_bounds__(256) void k_stats(const float* __restrict__ ema_cs,
                                               const float* __restrict__ cs,
                                               float* __restrict__ out_necs,
                                               float* __restrict__ n_acc) {
    __shared__ float red[256];
    const int t = threadIdx.x;
    const int k = blockIdx.x * 256 + t;
    float v = ema_cs[k] * DECAY + cs[k] * ONE_M_D;
    out_necs[k] = v;
    red[t] = v;
    __syncthreads();
    for (int o = 128; o > 0; o >>= 1) {
        if (t < o) red[t] += red[t + o];
        __syncthreads();
    }
    if (t == 0) atomicAdd(n_acc, red[0]);
}

// =====================================================================
// K4: scalar loss
// =====================================================================
__global__ void k_loss(const float* __restrict__ loss_acc, float* __restrict__ out_loss) {
    if (threadIdx.x == 0 && blockIdx.x == 0)
        out_loss[0] = COMMIT * loss_acc[0] / (float)((size_t)NTOK * D);
}

// =====================================================================
// K5: new_ema_w ([D,K]) and new_codebook ([K,D]) via LDS tile transpose.
// grid: (K/64)*(D/64) blocks x 256 threads
// =====================================================================
__global__ __launch_bounds__(256) void k_final(const float* __restrict__ ema_w,
                                               const float* __restrict__ dw,
                                               const float* __restrict__ necs,
                                               const float* __restrict__ n_acc,
                                               float* __restrict__ out_emaw,
                                               float* __restrict__ out_cb) {
    __shared__ float sdw[64][65];   // [k][d]
    __shared__ float sew[64][65];   // [d][k]
    __shared__ float scs[64];
    const int t  = threadIdx.x;
    const int k0 = (blockIdx.x & 127) * 64;   // K/64 = 128
    const int d0 = (blockIdx.x >> 7) * 64;    // D/64 = 16
    const float n = *n_acc;
    if (t < 64) {
        float c = necs[k0 + t];
        scs[t] = (c + EPSV) / (n + (float)K * EPSV) * n;
    }
    for (int j = 0; j < 16; ++j) {
        int e  = t + j * 256;
        int kk = e >> 6, dd = e & 63;         // coalesced dw read (d contiguous)
        sdw[kk][dd] = dw[(size_t)(k0 + kk) * D + d0 + dd];
        int dd2 = e >> 6, kk2 = e & 63;       // coalesced ema_w read (k contiguous)
        sew[dd2][kk2] = ema_w[(size_t)(d0 + dd2) * K + k0 + kk2];
    }
    __syncthreads();
    for (int j = 0; j < 16; ++j) {
        int e = t + j * 256;
        {   // new_ema_w, [D,K] layout, coalesced in k
            int dd = e >> 6, kk = e & 63;
            float nw = sew[dd][kk] * DECAY + sdw[kk][dd] * ONE_M_D;
            out_emaw[(size_t)(d0 + dd) * K + k0 + kk] = nw;
        }
        {   // new_codebook, [K,D] layout, coalesced in d
            int kk = e >> 6, dd = e & 63;
            float nw = sew[dd][kk] * DECAY + sdw[kk][dd] * ONE_M_D;
            out_cb[(size_t)(k0 + kk) * D + d0 + dd] = nw / scs[kk];
        }
    }
}

// =====================================================================
// host-side launcher
// =====================================================================
extern "C" void kernel_launch(void* const* d_in, const int* in_sizes, int n_in,
                              void* d_out, int out_size, void* d_ws, size_t ws_size,
                              hipStream_t stream) {
    (void)in_sizes; (void)n_in; (void)out_size; (void)ws_size;

    const float* z      = (const float*)d_in[0];   // [N, D]
    const float* cb     = (const float*)d_in[1];   // [K, D]
    const float* ema_cs = (const float*)d_in[2];   // [K]
    const float* ema_w  = (const float*)d_in[3];   // [D, K]

    float* out    = (float*)d_out;
    float* o_zq   = out;                           // 33,554,432
    float* o_ids  = o_zq   + (size_t)NTOK * D;     //     32,768
    float* o_loss = o_ids  + NTOK;                 //          1
    float* o_cb   = o_loss + 1;                    //  8,388,608
    float* o_necs = o_cb   + (size_t)K * D;        //      8,192
    float* o_emaw = o_necs + K;                    //  8,388,608

    char* ws = (char*)d_ws;
    unsigned short* w_c2   = (unsigned short*)(ws);                         // 32 MB: [2][K][D] bf16
    float*          w_cn   = (float*)(ws + (size_t)33554432);               // 32 KB
    float*          w_dw   = (float*)(ws + (size_t)33587200);               // 32 MB
    float*          w_cs   = (float*)(ws + (size_t)67141632);               // 32 KB
    int*            w_ids  = (int*)  (ws + (size_t)67174400);               // 128 KB
    float*          w_loss = (float*)(ws + (size_t)67305472);               // 4 B
    float*          w_n    = (float*)(ws + (size_t)67305476);               // 4 B

    // K0: prep codebook (bf16 hi/lo planes + ||c||^2), zero dw/cs/scalars
    k_prep_cb<<<K, 256, 0, stream>>>(cb, w_c2, w_cn);
    {
        int nz = K * D + K;                       // dw + cs contiguous
        k_zero<<<(nz + 255) / 256, 256, 0, stream>>>(w_dw, nz);
        k_zero<<<1, 256, 0, stream>>>(w_loss, 2); // loss + n
    }

    // K1: fused distance GEMM + argmin (WMMA bf16 hi/lo x3, TDM-fed tiles)
    k_argmin<<<NTOK / MT, 256, 0, stream>>>(z, w_c2, w_cn, w_ids, o_ids);

    // K2: gather z_q, loss partials, histogram, dw scatter
    k_gather<<<NTOK, 256, 0, stream>>>(z, cb, w_ids, o_zq, w_dw, w_cs, w_loss);

    // K3: EMA cluster sizes + n
    k_stats<<<K / 256, 256, 0, stream>>>(ema_cs, w_cs, o_necs, w_n);

    // K4: loss scalar
    k_loss<<<1, 1, 0, stream>>>(w_loss, o_loss);

    // K5: new_ema_w + new_codebook (tiled transpose)
    k_final<<<(K / 64) * (D / 64), 256, 0, stream>>>(ema_w, w_dw, o_necs, w_n,
                                                     o_emaw, o_cb);
}
